// SS2D_Block_24532853194990
// MI455X (gfx1250) — matrix operand order repaired and checked by reading, not compiled, not run on hardware
//
#include <hip/hip_runtime.h>
#include <hip/hip_bf16.h>
#include <math.h>

// ---------------- problem constants ----------------
#define BATCH   4
#define DMODEL  96
#define DINNER  192
#define DSTATE  16
#define DTRANK  6
#define KDIR    3
#define HIN     128
#define WIN     128
#define HO      64
#define WO      64
#define LSEQ    4096            // HO*WO
#define CPROJ   38              // DTRANK + 2*DSTATE
#define MPIX    (BATCH*LSEQ)    // 16384 rows for the big GEMMs

typedef __attribute__((ext_vector_type(16))) __bf16 v16bf;
typedef __attribute__((ext_vector_type(8)))  float  v8f;

union Frag { v16bf h; unsigned int u[8]; };

__device__ __forceinline__ float sigmoidf_(float x) { return 1.0f / (1.0f + __expf(-x)); }

// ---------------- 1) depthwise 7x7 stride-2 pad-3 (NHWC) ----------------
__global__ void k_dwconv7(const float* __restrict__ x, const float* __restrict__ w,
                          const float* __restrict__ bias, float* __restrict__ out) {
  int idx = blockIdx.x * blockDim.x + threadIdx.x;
  if (idx >= BATCH * HO * WO * DMODEL) return;
  int c = idx % DMODEL;
  int t = idx / DMODEL;
  int wx = t % WO; t /= WO;
  int wy = t % HO; int b = t / HO;
  float acc = bias[c];
  const float* wp = w + c * 49;
#pragma unroll
  for (int ky = 0; ky < 7; ++ky) {
    int iy = wy * 2 - 3 + ky;
    if ((unsigned)iy >= (unsigned)HIN) continue;
#pragma unroll
    for (int kx = 0; kx < 7; ++kx) {
      int ix = wx * 2 - 3 + kx;
      if ((unsigned)ix >= (unsigned)WIN) continue;
      acc += wp[ky * 7 + kx] * x[(((b * HIN + iy) * WIN + ix) * DMODEL) + c];
    }
  }
  out[idx] = acc;   // (B,64,64,96) NHWC == row-major (MPIX x 96)
}

// ---------------- 2/8) bf16 WMMA GEMM: out = A(MxK) * W(NxK)^T ----------------
// Block: 128 threads = 4 waves. Each block computes a 64(M) x 32(N) tile; each
// wave owns 16(M) x 32(N) = two 16x16 WMMA tiles sharing one A fragment.
// K is a compile-time constant so the chunk loop fully unrolls.
// Column-split write: n < split -> out0 (ld0), else out1 (ld1).
template<int KD>
__global__ __launch_bounds__(128)
void k_gemm_wmma(const float* __restrict__ A, const float* __restrict__ W,
                 float* __restrict__ out0, float* __restrict__ out1,
                 int split, int ld0, int ld1) {
  __shared__ __bf16 sA[64][32];   // [m][k]
  __shared__ __bf16 sB[32][32];   // [n][k]  (transposed so K-pairs are dword-contiguous)

  int m0 = blockIdx.x * 64;
  int n0 = blockIdx.y * 32;
  int tid  = threadIdx.x;
  int wave = tid >> 5;
  int lane = tid & 31;

  v8f acc0 = {0.f, 0.f, 0.f, 0.f, 0.f, 0.f, 0.f, 0.f};
  v8f acc1 = {0.f, 0.f, 0.f, 0.f, 0.f, 0.f, 0.f, 0.f};

  // per-lane fragment source rows (CDNA5 ISA 7.12.2, wave32)
  int mrow = (wave << 4) + (lane & 15);
  int kwA  = (lane < 16) ? 0 : 8;   // dword offset: A lanes 0-15 K0-15, lanes 16-31 K16-31
  int kwB  = (lane < 16) ? 0 : 8;   // dword offset: B lanes 0-15 K0-15, lanes 16-31 K16-31
  int nn   = lane & 15;
  const unsigned int* arow  = (const unsigned int*)&sA[mrow][0];
  const unsigned int* brow0 = (const unsigned int*)&sB[nn][0];
  const unsigned int* brow1 = (const unsigned int*)&sB[16 + nn][0];

#pragma unroll
  for (int kc = 0; kc < KD; kc += 32) {
    // stage A tile (64x32) -> bf16
    {
      int row = tid >> 1;
      int cb  = (tid & 1) * 16;
      const float* ap = A + (m0 + row) * KD + kc + cb;
#pragma unroll
      for (int j = 0; j < 16; ++j) sA[row][cb + j] = (__bf16)ap[j];
    }
    // stage B tile transposed: sB[n][k] = W[(n0+n)*KD + kc + k]
    {
      int n  = tid >> 2;
      int kb = (tid & 3) * 8;
      const float* wp = W + (n0 + n) * KD + kc + kb;
#pragma unroll
      for (int j = 0; j < 8; ++j) sB[n][kb + j] = (__bf16)wp[j];
    }
    __syncthreads();

    Frag af, bf0, bf1;
#pragma unroll
    for (int i = 0; i < 4; ++i) {           // dword loads: K pairs
      af.u[i]     = arow[kwA + i];          // K = kbA..kbA+7
      af.u[4 + i] = arow[4 + kwA + i];      // K = kbA+8..kbA+15
    }
#pragma unroll
    for (int v = 0; v < 8; ++v) {
      bf0.u[v] = brow0[kwB + v];
      bf1.u[v] = brow1[kwB + v];
    }
    acc0 = __builtin_amdgcn_wmma_f32_16x16x32_bf16(false, af.h, false, bf0.h,
                                                   (short)0, acc0, false, false);
    acc1 = __builtin_amdgcn_wmma_f32_16x16x32_bf16(false, af.h, false, bf1.h,
                                                   (short)0, acc1, false, false);
    __syncthreads();
  }

  // writeback: C/D layout — lane 0-15: M=r, N=lane; lane 16-31: M=8+r, N=lane-16
  int mbase = m0 + (wave << 4) + ((lane < 16) ? 0 : 8);
#pragma unroll
  for (int tile = 0; tile < 2; ++tile) {
    int nglob = n0 + tile * 16 + nn;
    const v8f& acc = tile ? acc1 : acc0;
    float* base; int ld;
    if (nglob < split) { base = out0; ld = ld0; }
    else               { base = out1; ld = ld1; nglob -= split; }
    float* p = base + (mbase * ld + nglob);   // 32-bit index math
#pragma unroll
    for (int r = 0; r < 8; ++r) p[r * ld] = acc[r];
  }
}

// ---------------- 3) depthwise 3x3 pad 1 + SiLU (NHWC, 192 ch) ----------------
__global__ void k_dwconv3_silu(const float* __restrict__ xi, const float* __restrict__ w,
                               const float* __restrict__ bias, float* __restrict__ out) {
  int idx = blockIdx.x * blockDim.x + threadIdx.x;
  if (idx >= BATCH * HO * WO * DINNER) return;
  int c = idx % DINNER;
  int t = idx / DINNER;
  int wx = t % WO; t /= WO;
  int wy = t % HO; int b = t / HO;
  float acc = bias[c];
  const float* wp = w + c * 9;
#pragma unroll
  for (int ky = 0; ky < 3; ++ky) {
    int iy = wy - 1 + ky;
    if ((unsigned)iy >= (unsigned)HO) continue;
#pragma unroll
    for (int kx = 0; kx < 3; ++kx) {
      int ix = wx - 1 + kx;
      if ((unsigned)ix >= (unsigned)WO) continue;
      acc += wp[ky * 3 + kx] * xi[((b * HO + iy) * WO + ix) * DINNER + c];
    }
  }
  out[idx] = acc * sigmoidf_(acc);
}

// direction map: l -> (hi, wi) for xs[k]
__device__ __forceinline__ void dir_map(int k, int l, int& hi, int& wi) {
  if (k == 0) { wi = l >> 6; hi = l & 63; }                                 // wh-major
  else if (k == 1) { int l2 = (LSEQ - 1) - l; hi = l2 >> 6; wi = l2 & 63; } // hw reversed
  else { int l2 = (LSEQ - 1) - l; wi = l2 >> 6; hi = l2 & 63; }             // wh reversed
}

// ---------------- 4) x_dbl[b,k,c,l] = sum_d xs[b,k,d,l] * xpw[k,c,d] ----------------
__global__ void k_xdbl(const float* __restrict__ xact, const float* __restrict__ xpw,
                       float* __restrict__ xdbl) {
  int idx = blockIdx.x * blockDim.x + threadIdx.x;
  if (idx >= BATCH * KDIR * CPROJ * LSEQ) return;
  int l = idx & (LSEQ - 1);
  int t = idx >> 12;
  int c = t % CPROJ; t /= CPROJ;
  int k = t % KDIR;  int b = t / KDIR;
  int hi, wi; dir_map(k, l, hi, wi);
  const float* xp = xact + ((b * HO + hi) * WO + wi) * DINNER;
  const float* wp = xpw + (k * CPROJ + c) * DINNER;
  float acc = 0.f;
#pragma unroll 8
  for (int d = 0; d < DINNER; ++d) acc += xp[d] * wp[d];
  xdbl[idx] = acc;                       // layout ((b*3+k)*38 + c)*4096 + l
}

// ---------------- 5) dts[b,k,d,l] = sum_r xdbl[b,k,r,l] * dtw[k,d,r] ----------------
__global__ void k_dtproj(const float* __restrict__ xdbl, const float* __restrict__ dtw,
                         float* __restrict__ dts) {
  int idx = blockIdx.x * blockDim.x + threadIdx.x;
  if (idx >= BATCH * KDIR * DINNER * LSEQ) return;
  int l = idx & (LSEQ - 1);
  int t = idx >> 12;
  int d = t % DINNER; t /= DINNER;
  int k = t % KDIR;   int b = t / KDIR;
  const float* xr = xdbl + ((b * KDIR + k) * CPROJ) * LSEQ + l;
  const float* wp = dtw + (k * DINNER + d) * DTRANK;
  float acc = 0.f;
#pragma unroll
  for (int r = 0; r < DTRANK; ++r) acc += xr[r * LSEQ] * wp[r];
  dts[idx] = acc;                        // layout ((b*3+k)*192 + d)*4096 + l
}

// ---------------- 6) selective scan: 16 lanes per (b,k,d), lane = state n ----------------
__global__ __launch_bounds__(256)
void k_scan(const float* __restrict__ xact, const float* __restrict__ xdbl,
            const float* __restrict__ dts, const float* __restrict__ A_logs,
            const float* __restrict__ dt_bias, const float* __restrict__ Dsv,
            float* __restrict__ oy) {
  int gt  = blockIdx.x * blockDim.x + threadIdx.x;
  int grp = gt >> 4;                 // one group of 16 lanes per (b,k,d)
  int n   = gt & 15;                 // state index
  if (grp >= BATCH * KDIR * DINNER) return;
  int d = grp % DINNER;
  int k = (grp / DINNER) % KDIR;
  int b = grp / (DINNER * KDIR);
  int kd = k * DINNER + d;

  float An   = -__expf(A_logs[kd * DSTATE + n]);
  float bias = dt_bias[kd];
  float Dv   = Dsv[kd];

  const float* dtp = dts  + ((b * KDIR + k) * DINNER + d) * LSEQ;
  const float* Bp  = xdbl + ((b * KDIR + k) * CPROJ + DTRANK + n) * LSEQ;
  const float* Cp  = xdbl + ((b * KDIR + k) * CPROJ + DTRANK + DSTATE + n) * LSEQ;
  const float* xa  = xact + b * LSEQ * DINNER + d;
  float*       op  = oy   + ((b * KDIR + k) * DINNER + d) * LSEQ;

  float h = 0.f;
  for (int l = 0; l < LSEQ; ++l) {
    if ((l & 63) == 0 && l + 256 < LSEQ) {        // stream hints -> global_prefetch
      __builtin_prefetch(Bp + l + 256, 0, 1);
      __builtin_prefetch(Cp + l + 256, 0, 1);
      __builtin_prefetch(dtp + l + 256, 0, 1);
    }
    int hi, wi; dir_map(k, l, hi, wi);
    float xv = xa[(hi * WO + wi) * DINNER];
    float tt = dtp[l] + bias;
    float delta = (tt > 20.f) ? tt : log1pf(__expf(tt));
    float Bv = Bp[l], Cv = Cp[l];
    h = h * __expf(delta * An) + (delta * xv) * Bv;
    float y = h * Cv;
    // reduce 16 states across the half-wave
    y += __shfl_xor(y, 8, 16);
    y += __shfl_xor(y, 4, 16);
    y += __shfl_xor(y, 2, 16);
    y += __shfl_xor(y, 1, 16);
    if (n == 0) op[l] = y + xv * Dv;
  }
}

// ---------------- 7) merge 3 directions + LayerNorm + SiLU gate ----------------
__global__ __launch_bounds__(192)
void k_merge_ln(const float* __restrict__ oy, const float* __restrict__ z,
                const float* __restrict__ gamma, const float* __restrict__ beta,
                float* __restrict__ yg) {
  __shared__ float s1[DINNER], s2[DINNER];
  int c   = threadIdx.x;
  int pix = blockIdx.x;              // b*4096 + l_hw
  int b   = pix >> 12;
  int l_hw = pix & (LSEQ - 1);
  int hi = l_hw >> 6, wi = l_hw & 63;
  int l_wh = wi * HO + hi;
  int bk = b * KDIR;
  float y = oy[((bk + 1) * DINNER + c) * LSEQ + (LSEQ - 1 - l_hw)]    // inv0
          + oy[((bk + 0) * DINNER + c) * LSEQ + l_wh]                 // wh_y (transposed)
          + oy[((bk + 2) * DINNER + c) * LSEQ + (LSEQ - 1 - l_wh)];   // invwh_y
  s1[c] = y; s2[c] = y * y;
  __syncthreads();
  for (int st = 96; st >= 3; st >>= 1) {
    if (c < st) { s1[c] += s1[c + st]; s2[c] += s2[c + st]; }
    __syncthreads();
  }
  float mu  = (s1[0] + s1[1] + s1[2]) * (1.f / DINNER);
  float var = (s2[0] + s2[1] + s2[2]) * (1.f / DINNER) - mu * mu;
  float yn = (y - mu) * rsqrtf(var + 1e-5f) * gamma[c] + beta[c];
  float zv = z[pix * DINNER + c];
  yg[pix * DINNER + c] = yn * (zv * sigmoidf_(zv));
}

// ---------------- 9) depthwise transposed 3x3 stride-2 -> (B,128,128,96) ----------------
__global__ void k_upconv(const float* __restrict__ o, const float* __restrict__ uw,
                         const float* __restrict__ ub, float* __restrict__ out) {
  int idx = blockIdx.x * blockDim.x + threadIdx.x;
  if (idx >= BATCH * HIN * WIN * DMODEL) return;
  int c = idx % DMODEL;
  int t = idx / DMODEL;
  int X = t % WIN; t /= WIN;
  int Y = t % HIN; int b = t / HIN;
  float acc = ub[c];
  const float* wp = uw + c * 9;
#pragma unroll
  for (int ty = 0; ty < 3; ++ty) {
    int py = Y + ty;
    if ((py & 1) == 0) continue;           // dilated input present only at odd positions
    int i = (py - 1) >> 1;
    if ((unsigned)i >= (unsigned)HO) continue;
#pragma unroll
    for (int tx = 0; tx < 3; ++tx) {
      int px = X + tx;
      if ((px & 1) == 0) continue;
      int j = (px - 1) >> 1;
      if ((unsigned)j >= (unsigned)WO) continue;
      acc += wp[(2 - ty) * 3 + (2 - tx)] * o[((b * HO + i) * WO + j) * DMODEL + c];
    }
  }
  out[idx] = acc;
}

// ---------------- launcher ----------------
extern "C" void kernel_launch(void* const* d_in, const int* in_sizes, int n_in,
                              void* d_out, int out_size, void* d_ws, size_t ws_size,
                              hipStream_t stream) {
  const float* x        = (const float*)d_in[0];
  const float* cb1_w    = (const float*)d_in[1];
  const float* cb1_b    = (const float*)d_in[2];
  const float* inproj_w = (const float*)d_in[3];
  const float* c2_w     = (const float*)d_in[4];
  const float* c2_b     = (const float*)d_in[5];
  const float* xpw      = (const float*)d_in[6];
  const float* dtw      = (const float*)d_in[7];
  const float* dtb      = (const float*)d_in[8];
  const float* A_logs   = (const float*)d_in[9];
  const float* Dsv      = (const float*)d_in[10];
  const float* gamma    = (const float*)d_in[11];
  const float* beta     = (const float*)d_in[12];
  const float* outp_w   = (const float*)d_in[13];
  const float* up_w     = (const float*)d_in[14];
  const float* up_b     = (const float*)d_in[15];
  float* out = (float*)d_out;

  float* ws = (float*)d_ws;
  // workspace layout (floats); xc buffer reused for out_proj result, xi buffer reused for gated y
  const size_t SZ_XC   = (size_t)MPIX * DMODEL;            // 1,572,864
  const size_t SZ_XI   = (size_t)MPIX * DINNER;            // 3,145,728
  const size_t SZ_XDBL = (size_t)BATCH * KDIR * CPROJ * LSEQ;
  const size_t SZ_DTS  = (size_t)BATCH * KDIR * DINNER * LSEQ;
  float* f_xc   = ws;
  float* f_xi   = f_xc + SZ_XC;
  float* f_z    = f_xi + SZ_XI;
  float* f_xact = f_z + SZ_XI;
  float* f_xdbl = f_xact + SZ_XI;
  float* f_dts  = f_xdbl + SZ_XDBL;
  float* f_oy   = f_dts + SZ_DTS;
  float* f_yg   = f_xi;   // reuse (xi dead after dwconv3)
  float* f_o    = f_xc;   // reuse (xc dead after in_proj)

  // 1) 7x7/s2 depthwise
  {
    int n = BATCH * HO * WO * DMODEL;
    k_dwconv7<<<(n + 255) / 256, 256, 0, stream>>>(x, cb1_w, cb1_b, f_xc);
  }
  // 2) in_proj GEMM (16384x96 -> 384), split into xi | z
  {
    dim3 grid(MPIX / 64, (2 * DINNER) / 32);
    k_gemm_wmma<DMODEL><<<grid, 128, 0, stream>>>(f_xc, inproj_w, f_xi, f_z,
                                                  DINNER, DINNER, DINNER);
  }
  // 3) 3x3 depthwise + SiLU
  {
    int n = BATCH * HO * WO * DINNER;
    k_dwconv3_silu<<<(n + 255) / 256, 256, 0, stream>>>(f_xi, c2_w, c2_b, f_xact);
  }
  // 4) per-direction projection -> x_dbl
  {
    int n = BATCH * KDIR * CPROJ * LSEQ;
    k_xdbl<<<(n + 255) / 256, 256, 0, stream>>>(f_xact, xpw, f_xdbl);
  }
  // 5) dt projection (6 -> 192)
  {
    int n = BATCH * KDIR * DINNER * LSEQ;
    k_dtproj<<<(n + 255) / 256, 256, 0, stream>>>(f_xdbl, dtw, f_dts);
  }
  // 6) selective scan
  {
    int n = BATCH * KDIR * DINNER * 16;    // 16 lanes per sequence
    k_scan<<<(n + 255) / 256, 256, 0, stream>>>(f_xact, f_xdbl, f_dts,
                                                A_logs, dtb, Dsv, f_oy);
  }
  // 7) merge directions + LayerNorm + gate
  {
    k_merge_ln<<<MPIX, DINNER, 0, stream>>>(f_oy, f_z, gamma, beta, f_yg);
  }
  // 8) out_proj GEMM (16384x192 -> 96)
  {
    dim3 grid(MPIX / 64, DMODEL / 32);
    k_gemm_wmma<DINNER><<<grid, 128, 0, stream>>>(f_yg, outp_w, f_o, f_o,
                                                  DMODEL, DMODEL, DMODEL);
  }
  // 9) transposed depthwise 3x3/s2 -> final output
  {
    int n = BATCH * HIN * WIN * DMODEL;
    k_upconv<<<(n + 255) / 256, 256, 0, stream>>>(f_o, up_w, up_b, out);
  }
  (void)in_sizes; (void)n_in; (void)out_size; (void)ws_size;
}
